// Network_4209067950298
// MI455X (gfx1250) — matrix operand order repaired and checked by reading, not compiled, not run on hardware
//
#include <hip/hip_runtime.h>

typedef __attribute__((ext_vector_type(16))) _Float16 v16h;
typedef __attribute__((ext_vector_type(8)))  float    v8f;

#define KSTEP 32
#define MTILE 64
#define NTILE 128

// ---------------------------------------------------------------------------
// f32 -> f16 pack
// ---------------------------------------------------------------------------
__global__ __launch_bounds__(256)
void pack_f16_kernel(const float* __restrict__ in, _Float16* __restrict__ out, long long n) {
  long long i = (long long)blockIdx.x * 256 + threadIdx.x;
  if (i < n) out[i] = (_Float16)in[i];
}

// ---------------------------------------------------------------------------
// 2x2 average pool, f16 [C,H,W] -> [C,H/2,W/2]
// ---------------------------------------------------------------------------
__global__ __launch_bounds__(256)
void avgpool2_kernel(const _Float16* __restrict__ in, _Float16* __restrict__ out,
                     int C, int H, int W) {
  int Ho = H >> 1, Wo = W >> 1;
  long long total = (long long)C * Ho * Wo;
  long long i = (long long)blockIdx.x * 256 + threadIdx.x;
  if (i >= total) return;
  int wo = (int)(i % Wo);
  long long t = i / Wo;
  int ho = (int)(t % Ho);
  int c  = (int)(t / Ho);
  const _Float16* p = in + ((long long)c * H + 2 * ho) * W + 2 * wo;
  float s = (float)p[0] + (float)p[1] + (float)p[W] + (float)p[W + 1];
  out[i] = (_Float16)(0.25f * s);
}

// ---------------------------------------------------------------------------
// bilinear 2x upsample, align_corners=True, f16 [C,Hi,Wi] -> [C,2Hi,2Wi]
// ---------------------------------------------------------------------------
__global__ __launch_bounds__(256)
void upsample2x_kernel(const _Float16* __restrict__ in, _Float16* __restrict__ out,
                       int C, int Hi, int Wi) {
  int Ho = 2 * Hi, Wo = 2 * Wi;
  long long total = (long long)C * Ho * Wo;
  long long i = (long long)blockIdx.x * 256 + threadIdx.x;
  if (i >= total) return;
  int x = (int)(i % Wo);
  long long t = i / Wo;
  int y = (int)(t % Ho);
  int c = (int)(t / Ho);
  float fy = (float)y * ((float)(Hi - 1) / (float)(Ho - 1));
  float fx = (float)x * ((float)(Wi - 1) / (float)(Wo - 1));
  int y0 = (int)fy, x0 = (int)fx;
  int y1 = (y0 + 1 < Hi) ? y0 + 1 : Hi - 1;
  int x1 = (x0 + 1 < Wi) ? x0 + 1 : Wi - 1;
  float wy = fy - (float)y0, wx = fx - (float)x0;
  const _Float16* p = in + (long long)c * Hi * Wi;
  float v00 = (float)p[(long long)y0 * Wi + x0];
  float v01 = (float)p[(long long)y0 * Wi + x1];
  float v10 = (float)p[(long long)y1 * Wi + x0];
  float v11 = (float)p[(long long)y1 * Wi + x1];
  float v = (1.f - wy) * ((1.f - wx) * v00 + wx * v01) +
            wy * ((1.f - wx) * v10 + wx * v11);
  out[i] = (_Float16)v;
}

// ---------------------------------------------------------------------------
// Implicit-GEMM 3x3 SAME conv via WMMA f16->f32.
//   Wgt  : [Cout, Cin*9]  f16   (A matrix)
//   Act  : [Cin, H*W]     f16   (im2col B on the fly)
//   Out  : [Cout, H*W]    f16   = act(W*B + bias) (+ resid)
// Block = 256 threads (8 wave32), tile = 64(M) x 128(N), K step = 32.
// Double-buffered LDS; staging = batch-issue all global loads (full MLP),
// then cndmask+ds_store, so the load pipe never drains per element.
// ---------------------------------------------------------------------------
__global__ __launch_bounds__(256)
void conv3x3_wmma_kernel(const _Float16* __restrict__ Wgt,
                         const _Float16* __restrict__ Act,
                         const float* __restrict__ Bias,
                         const _Float16* __restrict__ Resid,
                         _Float16* __restrict__ Out,
                         int Cout, int Cin, int H, int W, int doRelu) {
  const int HW = H * W;
  const int K  = Cin * 9;

  __shared__ _Float16 Asl[2][MTILE * KSTEP];   // [m][k]
  __shared__ _Float16 Bsl[2][NTILE * KSTEP];   // [n][k] (transposed tile)

  const int tid  = threadIdx.x;
  const int lane = tid & 31;
  const int wave = tid >> 5;
  const int wm   = wave & 3;     // 16-row M sub-tile
  const int wn   = wave >> 2;    // 64-col N half
  const int tileN = blockIdx.x * NTILE;
  const int tileM = blockIdx.y * MTILE;

  v8f acc[4];
  for (int i = 0; i < 4; ++i)
    for (int r = 0; r < 8; ++r) acc[i][r] = 0.0f;

  // B-load mapping: thread -> (n, 16-wide k range)
  const int  bn  = tid & 127;
  const int  bk0 = (tid >> 7) * 16;
  const int  ng  = tileN + bn;
  const int  pyv = ng / W;
  const int  pxv = ng - pyv * W;
  const bool nok = ng < HW;

  // ---- staging: phase 1 = issue ALL loads, phase 2 = select + LDS store ----
  auto stageA = [&](int buf, int k0) {
    _Float16 va[8];
    unsigned okbits = 0;
#pragma unroll
    for (int i = 0; i < 8; ++i) {
      int hidx = tid + i * 256;
      int m  = hidx >> 5;
      int k  = hidx & 31;
      int gm = tileM + m;
      int gk = k0 + k;
      bool ok = (gm < Cout) && (gk < K);
      okbits |= (ok ? 1u : 0u) << i;
      long long o = ok ? ((long long)gm * K + gk) : 0;   // clamp addr, always load
      va[i] = Wgt[o];
    }
#pragma unroll
    for (int i = 0; i < 8; ++i) {
      int hidx = tid + i * 256;
      int m = hidx >> 5;
      int k = hidx & 31;
      Asl[buf][m * KSTEP + k] = ((okbits >> i) & 1u) ? va[i] : (_Float16)0.f;
    }
  };
  auto stageB = [&](int buf, int k0) {
    _Float16 vb[16];
    unsigned okbits = 0;
#pragma unroll
    for (int kk = 0; kk < 16; ++kk) {
      int k  = bk0 + kk;
      int gk = k0 + k;
      unsigned ugk = (unsigned)gk;
      unsigned ci  = ugk / 9u;
      unsigned tap = ugk - ci * 9u;
      unsigned q   = tap / 3u;
      int yy = pyv + (int)q - 1;
      int xx = pxv + (int)(tap - q * 3u) - 1;
      bool ok = (gk < K) && nok &&
                ((unsigned)yy < (unsigned)H) && ((unsigned)xx < (unsigned)W);
      okbits |= (ok ? 1u : 0u) << kk;
      long long o = ok ? ((long long)ci * HW + (long long)yy * W + xx) : 0;
      vb[kk] = Act[o];                                   // unconditional load
    }
#pragma unroll
    for (int kk = 0; kk < 16; ++kk) {
      Bsl[buf][bn * KSTEP + bk0 + kk] =
          ((okbits >> kk) & 1u) ? vb[kk] : (_Float16)0.f;
    }
  };

  stageA(0, 0);
  stageB(0, 0);
  __syncthreads();

  int buf = 0;
  for (int k0 = 0; k0 < K; k0 += KSTEP) {
    const int k0n = k0 + KSTEP;
    if (k0n < K) {
      // speculative prefetch of the next weight rows (gfx1250 global_prefetch)
      __builtin_prefetch(Wgt + (long long)(tileM + (tid >> 2)) * K + k0n, 0, 1);
      stageA(buf ^ 1, k0n);
      stageB(buf ^ 1, k0n);
    }

    // --- A fragment (16x32 f16 per wave): ISA layout ---
    union { v16h h; unsigned u[8]; } afrag;
    {
      const unsigned* A32 = (const unsigned*)&Asl[buf][0];
      int g  = lane >> 4;
      int ml = (lane & 15) + wm * 16;
#pragma unroll
      for (int i = 0; i < 8; ++i) {
        int t = ((i < 4) ? 0 : 8) + g * 4 + (i & 3);   // K-pair index
        afrag.u[i] = A32[ml * (KSTEP / 2) + t];
      }
    }
    // --- 4 x (B fragment + WMMA) ---
    const unsigned* B32 = (const unsigned*)&Bsl[buf][0];
#pragma unroll
    for (int nb = 0; nb < 4; ++nb) {
      union { v16h h; unsigned u[8]; } bfrag;
      int g  = lane >> 4;
      int nl = (lane & 15) + wn * 64 + nb * 16;
#pragma unroll
      for (int v = 0; v < 8; ++v) {
        int kpair = ((v < 4) ? 0 : 8) + g * 4 + (v & 3); // K = 2*kpair (+1)
        bfrag.u[v] = B32[nl * (KSTEP / 2) + kpair];
      }
      acc[nb] = __builtin_amdgcn_wmma_f32_16x16x32_f16(
          false, afrag.h, false, bfrag.h, (short)0, acc[nb], false, false);
    }
    __syncthreads();
    buf ^= 1;
  }

  // --- epilogue: bias (+ relu) (+ residual), store f16 ---
  int g   = lane >> 4;
  int nl0 = lane & 15;
#pragma unroll
  for (int nb = 0; nb < 4; ++nb) {
#pragma unroll
    for (int r = 0; r < 8; ++r) {
      int m = tileM + wm * 16 + r + g * 8;
      int n = tileN + wn * 64 + nb * 16 + nl0;
      if (m < Cout && n < HW) {
        float val = acc[nb][r] + Bias[m];
        if (doRelu) val = val > 0.f ? val : 0.f;
        if (Resid) val += (float)Resid[(long long)m * HW + n];
        Out[(long long)m * HW + n] = (_Float16)val;
      }
    }
  }
}

// ---------------------------------------------------------------------------
// SepConv: out[c,y,x] (+)= sum_{i,j} pad_edge(img)[c,y+i,x+j] * V[i,y,x] * H[j,y,x]
// K=51, P=25.  16x16 pixel tile per block, per-channel image tile in LDS.
// ---------------------------------------------------------------------------
#define SCK 51
#define SCP 25
#define SCT 16
#define SCTW (SCT + SCK - 1)   // 66

__global__ __launch_bounds__(256)
void sepconv_kernel(const float* __restrict__ img,
                    const _Float16* __restrict__ Vk,
                    const _Float16* __restrict__ Hk,
                    float* __restrict__ out,
                    int H, int W, int addFlag) {
  __shared__ float tile[SCTW * SCTW];
  const int c  = blockIdx.z;
  const int bx = blockIdx.x * SCT;
  const int by = blockIdx.y * SCT;
  const long long HWll = (long long)H * W;
  const float* ip = img + (long long)c * HWll;

  for (int idx = threadIdx.x; idx < SCTW * SCTW; idx += 256) {
    int ty = idx / SCTW, tx = idx - ty * SCTW;
    int gy = by + ty - SCP;
    int gx = bx + tx - SCP;
    gy = gy < 0 ? 0 : (gy >= H ? H - 1 : gy);
    gx = gx < 0 ? 0 : (gx >= W ? W - 1 : gx);
    tile[idx] = ip[(long long)gy * W + gx];
  }
  __syncthreads();

  int lx = threadIdx.x & 15, ly = threadIdx.x >> 4;
  int gx = bx + lx, gy = by + ly;
  if (gx >= W || gy >= H) return;
  long long pix = (long long)gy * W + gx;

  float hk[SCK];
#pragma unroll
  for (int j = 0; j < SCK; ++j) hk[j] = (float)Hk[(long long)j * HWll + pix];

  float accv = 0.f;
  for (int i = 0; i < SCK; ++i) {
    const float* row = &tile[(ly + i) * SCTW + lx];
    float s = 0.f;
#pragma unroll
    for (int j = 0; j < SCK; ++j) s += row[j] * hk[j];
    accv += (float)Vk[(long long)i * HWll + pix] * s;
  }
  float* op = out + (long long)c * HWll + pix;
  if (addFlag) *op += accv; else *op = accv;
}

// ---------------------------------------------------------------------------
// Host orchestration
// ---------------------------------------------------------------------------
extern "C" void kernel_launch(void* const* d_in, const int* in_sizes, int n_in,
                              void* d_out, int out_size, void* d_ws, size_t ws_size,
                              hipStream_t stream) {
  (void)in_sizes; (void)n_in; (void)out_size; (void)ws_size;

  const int H0 = 512, W0 = 960;
  const int H1 = 256, W1 = 480;
  const int H2 = 128, W2 = 240;
  const int H3 = 64,  W3 = 120;
  const int H4 = 32,  W4 = 60;
  const int H5 = 16,  W5 = 30;
  const long long HW0 = (long long)H0 * W0;
  const long long HW1 = (long long)H1 * W1;
  const long long HW2 = (long long)H2 * W2;
  const long long HW3 = (long long)H3 * W3;
  const long long HW4 = (long long)H4 * W4;
  const long long HW5 = (long long)H5 * W5;

  // bump allocator over d_ws (deterministic sequence every call)
  size_t off = 0;
  auto alloc = [&](size_t bytes) -> void* {
    bytes = (bytes + 255) & ~(size_t)255;
    void* p = (char*)d_ws + off;
    off += bytes;
    return p;
  };
  auto allocH = [&](long long elems) -> _Float16* {
    return (_Float16*)alloc((size_t)elems * sizeof(_Float16));
  };

  auto pack = [&](const float* src, _Float16* dst, long long n) {
    unsigned blocks = (unsigned)((n + 255) / 256);
    pack_f16_kernel<<<dim3(blocks), 256, 0, stream>>>(src, dst, n);
  };

  struct CP { const _Float16* w; const float* b; int cin, cout; };
  auto packConv = [&](int i, int cin, int cout) -> CP {
    long long n = (long long)cout * cin * 9;
    _Float16* w = allocH(n);
    pack((const float*)d_in[i], w, n);
    CP c; c.w = w; c.b = (const float*)d_in[i + 1]; c.cin = cin; c.cout = cout;
    return c;
  };

  auto conv = [&](const CP& c, const _Float16* act, _Float16* outp,
                  int H, int W, int relu, const _Float16* resid) {
    int HW = H * W;
    dim3 grid((HW + NTILE - 1) / NTILE, (c.cout + MTILE - 1) / MTILE);
    conv3x3_wmma_kernel<<<grid, 256, 0, stream>>>(
        c.w, act, c.b, resid, outp, c.cout, c.cin, H, W, relu);
  };
  auto pool = [&](const _Float16* in, _Float16* outp, int C, int H, int W) {
    long long n = (long long)C * (H / 2) * (W / 2);
    avgpool2_kernel<<<dim3((unsigned)((n + 255) / 256)), 256, 0, stream>>>(in, outp, C, H, W);
  };
  auto upx2 = [&](const _Float16* in, _Float16* outp, int C, int Hi, int Wi) {
    long long n = (long long)C * 4 * Hi * Wi;
    upsample2x_kernel<<<dim3((unsigned)((n + 255) / 256)), 256, 0, stream>>>(in, outp, C, Hi, Wi);
  };

  // ---- d_in layout: [0]=first, [1]=second, then params sorted by key ----
  int idx = 2;
  auto take = [&](int n) { int r = idx; idx += n; return r; };
  const int conv1_   = take(6), conv2_ = take(6), conv3_ = take(6);
  const int conv4_   = take(6), conv5_ = take(6);
  const int deconv2_ = take(6), deconv3_ = take(6), deconv4_ = take(6), deconv5_ = take(6);
  const int horz1_   = take(8), horz2_ = take(8);
  const int up2_     = take(2), up3_ = take(2), up4_ = take(2), up5_ = take(2);
  const int vert1_   = take(8), vert2_ = take(8);

  auto packBasic = [&](int base, int cin, int cout, CP out3[3]) {
    out3[0] = packConv(base + 0, cin,  cout);
    out3[1] = packConv(base + 2, cout, cout);
    out3[2] = packConv(base + 4, cout, cout);
  };
  auto packSubnet = [&](int base, CP out4[4]) {
    out4[0] = packConv(base + 0, 64, 64);
    out4[1] = packConv(base + 2, 64, 64);
    out4[2] = packConv(base + 4, 64, 51);
    out4[3] = packConv(base + 6, 51, 51);
  };

  CP b1[3], b2[3], b3[3], b4[3], b5[3];
  CP dd5[3], dd4[3], dd3[3], dd2[3];
  CP sv1[4], sv2[4], sh1[4], sh2[4];
  packBasic(conv1_, 6, 32, b1);
  packBasic(conv2_, 32, 64, b2);
  packBasic(conv3_, 64, 128, b3);
  packBasic(conv4_, 128, 256, b4);
  packBasic(conv5_, 256, 512, b5);
  packBasic(deconv5_, 512, 512, dd5);
  packBasic(deconv4_, 512, 256, dd4);
  packBasic(deconv3_, 256, 128, dd3);
  packBasic(deconv2_, 128, 64, dd2);
  CP cup5 = packConv(up5_, 512, 512);
  CP cup4 = packConv(up4_, 256, 256);
  CP cup3 = packConv(up3_, 128, 128);
  CP cup2 = packConv(up2_, 64, 64);
  packSubnet(vert1_, sv1);
  packSubnet(vert2_, sv2);
  packSubnet(horz1_, sh1);
  packSubnet(horz2_, sh2);

  // ---- activation buffers ----
  _Float16* XIN  = allocH(6 * HW0);
  _Float16* A    = allocH(51 * HW0);
  _Float16* Bb   = allocH(51 * HW0);
  _Float16* C2   = allocH(64 * HW1);
  _Float16* C3   = allocH(128 * HW2);
  _Float16* C4   = allocH(256 * HW3);
  _Float16* C5   = allocH(512 * HW4);
  _Float16* P1   = allocH(32 * HW1);
  _Float16* P2   = allocH(64 * HW2);
  _Float16* P3   = allocH(128 * HW3);
  _Float16* P4   = allocH(256 * HW4);
  _Float16* P5   = allocH(512 * HW5);
  _Float16* COMB = allocH(64 * HW1);
  _Float16* VB   = allocH(51 * HW0);
  _Float16* HB   = allocH(51 * HW0);

  const float* firstImg  = (const float*)d_in[0];
  const float* secondImg = (const float*)d_in[1];
  float* outF = (float*)d_out;

  // ---- inputs to f16 (channel concat) ----
  pack(firstImg,  XIN,            3 * HW0);
  pack(secondImg, XIN + 3 * HW0,  3 * HW0);

  // ---- encoder ----
  conv(b1[0], XIN, A, H0, W0, 1, nullptr);
  conv(b1[1], A, Bb, H0, W0, 1, nullptr);
  conv(b1[2], Bb, A, H0, W0, 1, nullptr);      // c1 = A
  pool(A, P1, 32, H0, W0);
  conv(b2[0], P1, A, H1, W1, 1, nullptr);
  conv(b2[1], A, Bb, H1, W1, 1, nullptr);
  conv(b2[2], Bb, C2, H1, W1, 1, nullptr);     // c2
  pool(C2, P2, 64, H1, W1);
  conv(b3[0], P2, A, H2, W2, 1, nullptr);
  conv(b3[1], A, Bb, H2, W2, 1, nullptr);
  conv(b3[2], Bb, C3, H2, W2, 1, nullptr);     // c3
  pool(C3, P3, 128, H2, W2);
  conv(b4[0], P3, A, H3, W3, 1, nullptr);
  conv(b4[1], A, Bb, H3, W3, 1, nullptr);
  conv(b4[2], Bb, C4, H3, W3, 1, nullptr);     // c4
  pool(C4, P4, 256, H3, W3);
  conv(b5[0], P4, A, H4, W4, 1, nullptr);
  conv(b5[1], A, Bb, H4, W4, 1, nullptr);
  conv(b5[2], Bb, C5, H4, W4, 1, nullptr);     // c5
  pool(C5, P5, 512, H4, W4);

  // ---- decoder ----
  conv(dd5[0], P5, A, H5, W5, 1, nullptr);
  conv(dd5[1], A, Bb, H5, W5, 1, nullptr);
  conv(dd5[2], Bb, A, H5, W5, 1, nullptr);     // d5 = A
  upx2(A, Bb, 512, H5, W5);
  conv(cup5, Bb, A, H4, W4, 1, C5);            // u5 = A

  conv(dd4[0], A, Bb, H4, W4, 1, nullptr);
  conv(dd4[1], Bb, A, H4, W4, 1, nullptr);
  conv(dd4[2], A, Bb, H4, W4, 1, nullptr);     // d4 = Bb
  upx2(Bb, A, 256, H4, W4);
  conv(cup4, A, Bb, H3, W3, 1, C4);            // u4 = Bb

  conv(dd3[0], Bb, A, H3, W3, 1, nullptr);
  conv(dd3[1], A, Bb, H3, W3, 1, nullptr);
  conv(dd3[2], Bb, A, H3, W3, 1, nullptr);     // d3 = A
  upx2(A, Bb, 128, H3, W3);
  conv(cup3, Bb, A, H2, W2, 1, C3);            // u3 = A

  conv(dd2[0], A, Bb, H2, W2, 1, nullptr);
  conv(dd2[1], Bb, A, H2, W2, 1, nullptr);
  conv(dd2[2], A, Bb, H2, W2, 1, nullptr);     // d2 = Bb
  upx2(Bb, A, 64, H2, W2);
  conv(cup2, A, COMB, H1, W1, 1, C2);          // comb

  // ---- subnets + sepconv ----
  auto subnet = [&](CP ps[4], _Float16* dst) {
    conv(ps[0], COMB, A, H1, W1, 1, nullptr);
    conv(ps[1], A, Bb, H1, W1, 1, nullptr);
    conv(ps[2], Bb, A, H1, W1, 1, nullptr);    // 51ch @ H1xW1
    upx2(A, Bb, 51, H1, W1);                   // 51ch @ H0xW0
    conv(ps[3], Bb, dst, H0, W0, 0, nullptr);  // no relu
  };

  dim3 scGrid(W0 / SCT, H0 / SCT, 3);

  subnet(sv1, VB);
  subnet(sh1, HB);
  sepconv_kernel<<<scGrid, 256, 0, stream>>>(firstImg, VB, HB, outF, H0, W0, 0);

  subnet(sv2, VB);
  subnet(sh2, HB);
  sepconv_kernel<<<scGrid, 256, 0, stream>>>(secondImg, VB, HB, outF, H0, W0, 1);
}
